// DeepGraphConv_Surv_68642167325076
// MI455X (gfx1250) — matrix operand order repaired and checked by reading, not compiled, not run on hardware
//
#include <hip/hip_runtime.h>
#include <hip/hip_bf16.h>

// ---------------------------------------------------------------------------
// CDNA5 (gfx1250) implementation of DeepGraphConv_Surv forward pass.
// GEMMs on v_wmma_f32_16x16x32_bf16 (f32 accumulate). Weights pre-converted
// to bf16 *transposed* so both A and B WMMA fragments are contiguous 16B LDS
// loads (ds_load_b128). Prefetch at near scope; fast exp-based activations.
// ---------------------------------------------------------------------------

typedef __bf16 bf16;
typedef bf16  v16bf  __attribute__((ext_vector_type(16)));
typedef bf16  bf16x8 __attribute__((ext_vector_type(8)));
typedef bf16  bf16x4 __attribute__((ext_vector_type(4)));
typedef float v8f    __attribute__((ext_vector_type(8)));

#define NDIM   50000
#define EDIM   800000
#define INDIM  1792
#define HDIM   128
#define H3DIM  384
#define GDIM   8

union frag_u { v16bf v; bf16x8 h[2]; };

// ---------------------------------------------------------------------------
// f32 [K x Nw] -> bf16 transposed [Nw x K]
// ---------------------------------------------------------------------------
__global__ void cvt_transpose_bf16(const float* __restrict__ src,
                                   bf16* __restrict__ dst, int K, int Nw) {
  int i = blockIdx.x * blockDim.x + threadIdx.x;
  if (i >= K * Nw) return;
  int k = i / Nw, n = i % Nw;
  dst[(size_t)n * K + k] = (bf16)src[i];
}

// ---------------------------------------------------------------------------
// Init: s = bc, gmax keys = INT_MIN, gsum = 0, pooled = 0
// ---------------------------------------------------------------------------
__global__ void init_bufs(float* __restrict__ s, int* __restrict__ gmaxk,
                          float* __restrict__ gsum, float* __restrict__ pooled,
                          const float* __restrict__ bc, int N, int G) {
  int i = blockIdx.x * blockDim.x + threadIdx.x;
  if (i < N) s[i] = bc[0];
  if (i < G) { gmaxk[i] = (int)0x80000000; gsum[i] = 0.0f; }
  if (i < G * H3DIM) pooled[i] = 0.0f;
}

// ---------------------------------------------------------------------------
// Fragment loaders: two 8-bf16 runs (k = 8h.., 16+8h..) -> two ds_load_b128
// ---------------------------------------------------------------------------
__device__ __forceinline__ v16bf load_afrag(const bf16* ap, int half) {
  frag_u u;
  u.h[0] = *(const bf16x8*)(ap + half * 8);
  u.h[1] = *(const bf16x8*)(ap + 16 + half * 8);
  return u.v;
}
__device__ __forceinline__ v16bf load_bfrag(const bf16* bp, int half) {
  frag_u u;
  u.h[0] = *(const bf16x8*)(bp + half * 8);
  u.h[1] = *(const bf16x8*)(bp + 16 + half * 8);
  return u.v;
}

// fast sigmoid / tanh sharing one exp pipeline
__device__ __forceinline__ float fast_sigmoid(float x) {
  return 1.0f / (1.0f + __expf(-x));
}
__device__ __forceinline__ float fast_tanh(float x) {
  return 1.0f - 2.0f / (1.0f + __expf(2.0f * x));
}

// ---------------------------------------------------------------------------
// Generic WMMA GEMM:  C[:, ccol0 + nblk : +128] = act(A[MxK] @ W + bias)
// W given transposed bf16 [Nw x K]. Block tile 128x128, K-step 32.
// 8 waves, each wave owns 2x4 16x16 tiles.
// ---------------------------------------------------------------------------
template <int RELU>
__global__ __launch_bounds__(256) void gemm_bf16_wmma(
    const float* __restrict__ A, int lda, int M, int K,
    const bf16* __restrict__ Wt,           // [Nw x K] transposed
    const float* __restrict__ bias,
    float* __restrict__ C, int ldc, int ccol0) {
  __shared__ bf16 As[128][40];   // 128 rows x 32 k (+8 pad)
  __shared__ bf16 Bs[128][40];   // 128 cols x 32 k (+8 pad), transposed tile

  const int tid    = threadIdx.x;
  const int wave   = tid >> 5;
  const int lane   = tid & 31;
  const int half   = lane >> 4;
  const int lr     = lane & 15;
  const int warp_m = wave & 3;    // 4 x 32 rows
  const int warp_n = wave >> 2;   // 2 x 64 cols
  const int m0     = blockIdx.x * 128;
  const int nblk   = blockIdx.y * 128;

  v8f acc[2][4];
#pragma unroll
  for (int i = 0; i < 2; ++i)
#pragma unroll
    for (int j = 0; j < 4; ++j) acc[i][j] = (v8f){0,0,0,0,0,0,0,0};

  for (int k0 = 0; k0 < K; k0 += 32) {
    // --- stage A tile (128x32 f32 -> bf16 LDS), 4 float4 per thread ---
#pragma unroll
    for (int it = 0; it < 4; ++it) {
      int fi  = tid + it * 256;          // 0..1023 float4 slots
      int row = fi >> 3;
      int k4  = (fi & 7) << 2;
      int gr  = m0 + row; if (gr > M - 1) gr = M - 1;
      const float* gp = A + (size_t)gr * lda + (k0 + k4);
      float4 v = *(const float4*)gp;
      if (k0 + 32 < K) __builtin_prefetch(gp + 32, 0, 3);  // near-scope
      bf16* p = &As[row][k4];
      p[0] = (bf16)v.x; p[1] = (bf16)v.y; p[2] = (bf16)v.z; p[3] = (bf16)v.w;
    }
    // --- stage W tile: Bs[col][k], contiguous bf16x4 both sides ---
#pragma unroll
    for (int it = 0; it < 4; ++it) {
      int wi  = tid + it * 256;          // 0..1023 bf16x4 slots
      int col = wi >> 3;
      int k4  = (wi & 7) << 2;
      *(bf16x4*)&Bs[col][k4] =
          *(const bf16x4*)(Wt + (size_t)(nblk + col) * K + k0 + k4);
    }
    __syncthreads();

    v16bf af[2], bfr[4];
#pragma unroll
    for (int i = 0; i < 2; ++i)
      af[i] = load_afrag(&As[warp_m * 32 + i * 16 + lr][0], half);
#pragma unroll
    for (int j = 0; j < 4; ++j)
      bfr[j] = load_bfrag(&Bs[warp_n * 64 + j * 16 + lr][0], half);
#pragma unroll
    for (int i = 0; i < 2; ++i)
#pragma unroll
      for (int j = 0; j < 4; ++j)
        acc[i][j] = __builtin_amdgcn_wmma_f32_16x16x32_bf16(
            false, af[i], false, bfr[j], (short)0, acc[i][j], false, false);
    __syncthreads();
  }

  // --- epilogue: bias + relu + store ---
#pragma unroll
  for (int i = 0; i < 2; ++i) {
#pragma unroll
    for (int j = 0; j < 4; ++j) {
      int col = warp_n * 64 + j * 16 + lr;
      float bv = bias[nblk + col];
#pragma unroll
      for (int v = 0; v < 8; ++v) {
        int grow = m0 + warp_m * 32 + i * 16 + half * 8 + v;
        if (grow < M) {
          float val = acc[i][j][v] + bv;
          if (RELU) val = fmaxf(val, 0.0f);
          C[(size_t)grow * ldc + ccol0 + nblk + col] = val;
        }
      }
    }
  }
}

// ---------------------------------------------------------------------------
// z[n,0:128] = hcol[n,0:128]   (hcol = hp + column offset, ld = 384)
// ---------------------------------------------------------------------------
__global__ void copy_h_to_z(const float* __restrict__ hcol, int ldh,
                            float* __restrict__ z, int n4) {
  int i = blockIdx.x * blockDim.x + threadIdx.x;
  if (i >= n4) return;
  int row = i >> 5;
  int c4  = (i & 31) << 2;
  *(float4*)(z + (size_t)row * HDIM + c4) =
      *(const float4*)(hcol + (size_t)row * ldh + c4);
}

// ---------------------------------------------------------------------------
// z[dst] += h[src] over all edges (L2-resident atomics)
// ---------------------------------------------------------------------------
__global__ void edge_agg(const float* __restrict__ hcol, int ldh,
                         const int* __restrict__ src,
                         const int* __restrict__ dst, int E,
                         float* __restrict__ z) {
  int idx = blockIdx.x * blockDim.x + threadIdx.x;
  if (idx >= E * 32) return;
  int e  = idx >> 5;
  int c4 = (idx & 31) << 2;
  int sn = src[e], dn = dst[e];
  float4 v = *(const float4*)(hcol + (size_t)sn * ldh + c4);
  float* zp = z + (size_t)dn * HDIM + c4;
  atomicAdd(zp + 0, v.x); atomicAdd(zp + 1, v.y);
  atomicAdd(zp + 2, v.z); atomicAdd(zp + 3, v.w);
}

// ---------------------------------------------------------------------------
// Fused attention: s[n] += sum_j tanh((hp@Wa+ba)_j) * sig((hp@Wb+bb)_j) * Wc_j
// 64x128 tile per block, dual WMMA accumulators, grid.y covers 384 cols.
// ---------------------------------------------------------------------------
__global__ __launch_bounds__(256) void attn_fused_wmma(
    const float* __restrict__ hp, int M,
    const bf16* __restrict__ Wat, const bf16* __restrict__ Wbt,
    const float* __restrict__ ba, const float* __restrict__ bb,
    const float* __restrict__ Wc, float* __restrict__ s) {
  __shared__ bf16 As[64][40];
  __shared__ bf16 Bsa[128][40];
  __shared__ bf16 Bsb[128][40];

  const int tid = threadIdx.x, wave = tid >> 5, lane = tid & 31;
  const int half = lane >> 4, lr = lane & 15;
  const int warp_m = wave & 1;   // 2 x 32 rows
  const int warp_n = wave >> 1;  // 4 x 32 cols
  const int m0   = blockIdx.x * 64;
  const int nblk = blockIdx.y * 128;

  v8f acca[2][2], accb[2][2];
#pragma unroll
  for (int i = 0; i < 2; ++i)
#pragma unroll
    for (int j = 0; j < 2; ++j) {
      acca[i][j] = (v8f){0,0,0,0,0,0,0,0};
      accb[i][j] = (v8f){0,0,0,0,0,0,0,0};
    }

  for (int k0 = 0; k0 < H3DIM; k0 += 32) {
#pragma unroll
    for (int it = 0; it < 2; ++it) {
      int fi  = tid + it * 256;        // 0..511 float4 slots (64x8)
      int row = fi >> 3;
      int k4  = (fi & 7) << 2;
      int gr  = m0 + row; if (gr > M - 1) gr = M - 1;
      const float* gp = hp + (size_t)gr * H3DIM + k0 + k4;
      float4 v = *(const float4*)gp;
      if (k0 + 32 < H3DIM) __builtin_prefetch(gp + 32, 0, 3);
      bf16* p = &As[row][k4];
      p[0] = (bf16)v.x; p[1] = (bf16)v.y; p[2] = (bf16)v.z; p[3] = (bf16)v.w;
    }
#pragma unroll
    for (int it = 0; it < 4; ++it) {
      int wi  = tid + it * 256;
      int col = wi >> 3;
      int k4  = (wi & 7) << 2;
      size_t gidx = (size_t)(nblk + col) * H3DIM + k0 + k4;
      *(bf16x4*)&Bsa[col][k4] = *(const bf16x4*)(Wat + gidx);
      *(bf16x4*)&Bsb[col][k4] = *(const bf16x4*)(Wbt + gidx);
    }
    __syncthreads();

    v16bf af[2], bfa[2], bfb[2];
#pragma unroll
    for (int i = 0; i < 2; ++i)
      af[i] = load_afrag(&As[warp_m * 32 + i * 16 + lr][0], half);
#pragma unroll
    for (int j = 0; j < 2; ++j) {
      bfa[j] = load_bfrag(&Bsa[warp_n * 32 + j * 16 + lr][0], half);
      bfb[j] = load_bfrag(&Bsb[warp_n * 32 + j * 16 + lr][0], half);
    }
#pragma unroll
    for (int i = 0; i < 2; ++i)
#pragma unroll
      for (int j = 0; j < 2; ++j) {
        acca[i][j] = __builtin_amdgcn_wmma_f32_16x16x32_bf16(
            false, af[i], false, bfa[j], (short)0, acca[i][j], false, false);
        accb[i][j] = __builtin_amdgcn_wmma_f32_16x16x32_bf16(
            false, af[i], false, bfb[j], (short)0, accb[i][j], false, false);
      }
    __syncthreads();
  }

  // epilogue: gate, scale by Wc, reduce 16 columns across lanes, atomicAdd
#pragma unroll
  for (int i = 0; i < 2; ++i) {
    float rowsum[8];
#pragma unroll
    for (int v = 0; v < 8; ++v) rowsum[v] = 0.0f;
#pragma unroll
    for (int j = 0; j < 2; ++j) {
      int gcol = nblk + warp_n * 32 + j * 16 + lr;
      float bav = ba[gcol], bbv = bb[gcol], wcv = Wc[gcol];
#pragma unroll
      for (int v = 0; v < 8; ++v) {
        float pa = fast_tanh(acca[i][j][v] + bav);
        float sg = fast_sigmoid(accb[i][j][v] + bbv);
        rowsum[v] += pa * sg * wcv;
      }
    }
#pragma unroll
    for (int v = 0; v < 8; ++v) {
      float t = rowsum[v];
      t += __shfl_xor(t, 1); t += __shfl_xor(t, 2);
      t += __shfl_xor(t, 4); t += __shfl_xor(t, 8);
      if (lr == 0) {
        int grow = m0 + warp_m * 32 + i * 16 + half * 8 + v;
        if (grow < M) atomicAdd(&s[grow], t);
      }
    }
  }
}

// ---------------------------------------------------------------------------
// Segment softmax helpers (monotonic int-key trick for float atomicMax)
// ---------------------------------------------------------------------------
__device__ __forceinline__ int f2key(float f) {
  int b = __float_as_int(f);
  return b ^ ((b >> 31) & 0x7fffffff);
}

__global__ void seg_max(const float* __restrict__ s,
                        const int* __restrict__ batch,
                        int* __restrict__ gmaxk, int N) {
  int i = blockIdx.x * blockDim.x + threadIdx.x;
  if (i < N) atomicMax(&gmaxk[batch[i]], f2key(s[i]));
}

__global__ void seg_expsum(float* __restrict__ s,
                           const int* __restrict__ batch,
                           const int* __restrict__ gmaxk,
                           float* __restrict__ gsum, int N) {
  int i = blockIdx.x * blockDim.x + threadIdx.x;
  if (i >= N) return;
  int g = batch[i];
  int k = gmaxk[g];
  float mx = __int_as_float(k ^ ((k >> 31) & 0x7fffffff));
  float e = __expf(s[i] - mx);
  s[i] = e;
  atomicAdd(&gsum[g], e);
}

// ---------------------------------------------------------------------------
// pooled[g,:] += hp[n,:] * (e[n]/gsum[g]); batch is sorted -> flush on change
// ---------------------------------------------------------------------------
__global__ void pool_kernel(const float* __restrict__ hp,
                            const float* __restrict__ e,
                            const float* __restrict__ gsum,
                            const int* __restrict__ batch,
                            float* __restrict__ pooled, int N) {
  int c  = threadIdx.x;
  int n0 = blockIdx.x * 64;
  float acc = 0.0f;
  int curg = -1;
  for (int t = 0; t < 64; ++t) {
    int n = n0 + t;
    if (n >= N) break;
    int g = batch[n];
    if (g != curg) {
      if (curg >= 0) atomicAdd(&pooled[curg * H3DIM + c], acc);
      curg = g; acc = 0.0f;
    }
    float w = e[n] / gsum[g];
    acc += hp[(size_t)n * H3DIM + c] * w;
  }
  if (curg >= 0) atomicAdd(&pooled[curg * H3DIM + c], acc);
}

// ---------------------------------------------------------------------------
// out[g] = relu(pooled[g]@Wr + br) @ Wk + bk   (tiny: 8 blocks x 384 threads)
// ---------------------------------------------------------------------------
__global__ void head_kernel(const float* __restrict__ pooled,
                            const float* __restrict__ Wr,
                            const float* __restrict__ br,
                            const float* __restrict__ Wk,
                            const float* __restrict__ bk,
                            float* __restrict__ out) {
  __shared__ float sp[H3DIM];
  __shared__ float red[H3DIM];
  int g = blockIdx.x, j = threadIdx.x;
  sp[j] = pooled[g * H3DIM + j];
  __syncthreads();
  float acc = br[j];
  for (int k = 0; k < H3DIM; ++k) acc += sp[k] * Wr[k * H3DIM + j];
  red[j] = fmaxf(acc, 0.0f) * Wk[j];
  __syncthreads();
  for (int off = 256; off > 0; off >>= 1) {
    if (j < off && j + off < H3DIM) red[j] += red[j + off];
    __syncthreads();
  }
  if (j == 0) out[g] = red[0] + bk[0];
}

// ---------------------------------------------------------------------------
// Host orchestration
// ---------------------------------------------------------------------------
extern "C" void kernel_launch(void* const* d_in, const int* in_sizes, int n_in,
                              void* d_out, int out_size, void* d_ws,
                              size_t ws_size, hipStream_t stream) {
  const float* x    = (const float*)d_in[0];
  const int*   eidx = (const int*)d_in[1];
  const int*   batch= (const int*)d_in[2];
  const float* W_fc = (const float*)d_in[3];  const float* b_fc = (const float*)d_in[4];
  const float* W1a  = (const float*)d_in[5];  const float* b1a  = (const float*)d_in[6];
  const float* W1b  = (const float*)d_in[7];  const float* b1b  = (const float*)d_in[8];
  const float* W2a  = (const float*)d_in[9];  const float* b2a  = (const float*)d_in[10];
  const float* W2b  = (const float*)d_in[11]; const float* b2b  = (const float*)d_in[12];
  const float* Wa   = (const float*)d_in[13]; const float* ba   = (const float*)d_in[14];
  const float* Wb   = (const float*)d_in[15]; const float* bb   = (const float*)d_in[16];
  const float* Wc   = (const float*)d_in[17]; const float* bc   = (const float*)d_in[18];
  const float* Wr   = (const float*)d_in[19]; const float* br   = (const float*)d_in[20];
  const float* Wk   = (const float*)d_in[21]; const float* bk   = (const float*)d_in[22];

  const int N = in_sizes[2];       // 50000
  const int E = in_sizes[1] / 2;   // 800000
  float* out = (float*)d_out;

  // ---- workspace layout (256B aligned) ----
  char* base = (char*)d_ws;
  size_t off = 0;
  auto alloc = [&](size_t bytes) {
    size_t o = off;
    off = (off + bytes + 255) & ~(size_t)255;
    return o;
  };
  bf16* bWfc = (bf16*)(base + alloc((size_t)INDIM * HDIM * 2));
  bf16* bW1a = (bf16*)(base + alloc((size_t)HDIM * HDIM * 2));
  bf16* bW1b = (bf16*)(base + alloc((size_t)HDIM * HDIM * 2));
  bf16* bW2a = (bf16*)(base + alloc((size_t)HDIM * HDIM * 2));
  bf16* bW2b = (bf16*)(base + alloc((size_t)HDIM * HDIM * 2));
  bf16* bWa  = (bf16*)(base + alloc((size_t)H3DIM * H3DIM * 2));
  bf16* bWb  = (bf16*)(base + alloc((size_t)H3DIM * H3DIM * 2));
  float* hp     = (float*)(base + alloc((size_t)N * H3DIM * 4));
  float* z      = (float*)(base + alloc((size_t)N * HDIM * 4));
  float* t      = (float*)(base + alloc((size_t)N * HDIM * 4));
  float* s      = (float*)(base + alloc((size_t)N * 4));
  int*   gmaxk  = (int*)  (base + alloc((size_t)GDIM * 4));
  float* gsum   = (float*)(base + alloc((size_t)GDIM * 4));
  float* pooled = (float*)(base + alloc((size_t)GDIM * H3DIM * 4));
  (void)ws_size; (void)n_in; (void)out_size;

  const int* esrc = eidx;
  const int* edst = eidx + E;

  // ---- weights to transposed bf16 ----
  auto cvt = [&](const float* src, bf16* dst, int K, int Nw) {
    int n = K * Nw;
    cvt_transpose_bf16<<<(n + 255) / 256, 256, 0, stream>>>(src, dst, K, Nw);
  };
  cvt(W_fc, bWfc, INDIM, HDIM);
  cvt(W1a, bW1a, HDIM, HDIM);  cvt(W1b, bW1b, HDIM, HDIM);
  cvt(W2a, bW2a, HDIM, HDIM);  cvt(W2b, bW2b, HDIM, HDIM);
  cvt(Wa, bWa, H3DIM, H3DIM);  cvt(Wb, bWb, H3DIM, H3DIM);

  init_bufs<<<(N + 255) / 256, 256, 0, stream>>>(s, gmaxk, gsum, pooled, bc, N, GDIM);

  const dim3 g128((N + 127) / 128, 1);
  const int n4 = N * 32;

  // h = relu(x @ W_fc + b_fc)  -> hp[:, 0:128]
  gemm_bf16_wmma<1><<<g128, 256, 0, stream>>>(x, INDIM, N, INDIM, bWfc,
                                              b_fc, hp, H3DIM, 0);

  // ---- GIN layer 1 ----
  copy_h_to_z<<<(n4 + 255) / 256, 256, 0, stream>>>(hp + 0, H3DIM, z, n4);
  edge_agg<<<(E * 32 + 255) / 256, 256, 0, stream>>>(hp + 0, H3DIM, esrc, edst, E, z);
  gemm_bf16_wmma<1><<<g128, 256, 0, stream>>>(z, HDIM, N, HDIM, bW1a,
                                              b1a, t, HDIM, 0);
  gemm_bf16_wmma<1><<<g128, 256, 0, stream>>>(t, HDIM, N, HDIM, bW1b,
                                              b1b, hp, H3DIM, HDIM);   // h1

  // ---- GIN layer 2 ----
  copy_h_to_z<<<(n4 + 255) / 256, 256, 0, stream>>>(hp + HDIM, H3DIM, z, n4);
  edge_agg<<<(E * 32 + 255) / 256, 256, 0, stream>>>(hp + HDIM, H3DIM, esrc, edst, E, z);
  gemm_bf16_wmma<1><<<g128, 256, 0, stream>>>(z, HDIM, N, HDIM, bW2a,
                                              b2a, t, HDIM, 0);
  gemm_bf16_wmma<1><<<g128, 256, 0, stream>>>(t, HDIM, N, HDIM, bW2b,
                                              b2b, hp, H3DIM, 2 * HDIM); // h2

  // ---- fused gated attention score: s[n] = bc + (tanh⊙sig) @ Wc ----
  attn_fused_wmma<<<dim3((N + 63) / 64, 3), 256, 0, stream>>>(
      hp, N, bWa, bWb, ba, bb, Wc, s);

  // ---- segment softmax + weighted pooling ----
  seg_max<<<(N + 255) / 256, 256, 0, stream>>>(s, batch, gmaxk, N);
  seg_expsum<<<(N + 255) / 256, 256, 0, stream>>>(s, batch, gmaxk, gsum, N);
  pool_kernel<<<(N + 63) / 64, 384, 0, stream>>>(hp, s, gsum, batch, pooled, N);

  // ---- head ----
  head_kernel<<<GDIM, 384, 0, stream>>>(pooled, Wr, br, Wk, bk, out);
}
